// CasualAttentionHead_60842506715526
// MI455X (gfx1250) — matrix operand order repaired and checked by reading, not compiled, not run on hardware
//
#include <hip/hip_runtime.h>

// ---------------------------------------------------------------------------
// Causal multi-head attention block for MI455X (gfx1250), wave32 + WMMA.
//   qkv = x @ Wqkv^T + bqkv      (f16 WMMA, f32 accum)
//   flash-attention per head     (f16 WMMA, f32 accum, online softmax,
//                                 32 query rows/wave, masked tail block only,
//                                 row-sums via P @ ones WMMA)
//   out = O @ Wproj^T + bproj    (f16 WMMA, f32 accum, f32 out)
// ---------------------------------------------------------------------------

typedef _Float16 f16;
typedef __attribute__((ext_vector_type(16))) _Float16 v16h;
typedef __attribute__((ext_vector_type(8)))  _Float16 v8h;
typedef __attribute__((ext_vector_type(8)))  float    v8f;

#define WMMA_F32_F16(a, b, c) \
  __builtin_amdgcn_wmma_f32_16x16x32_f16(false, (a), false, (b), (short)0, (c), false, false)

static constexpr int B_  = 2;
static constexpr int T_  = 2048;
static constexpr int C_  = 1024;
static constexpr int H_  = 16;
static constexpr int D_  = 64;
static constexpr int C3_ = 3 * C_;

// A-operand fragment (16x32 f16). Per ISA 16-bit A layout:
// lanes 0-15 hold K {k..k+7, k+16..k+23}; lanes 16-31 hold K {k+8..k+15, k+24..k+31}.
__device__ __forceinline__ v16h load_frag_a(const f16* __restrict__ base, int ld,
                                            int row, int k, int hi) {
  const f16* p = base + (size_t)row * ld + k + hi * 8;
  v8h lo = *(const v8h*)p;
  v8h hh = *(const v8h*)(p + 16);
  v16h r;
#pragma unroll
  for (int i = 0; i < 8; i++) { r[i] = lo[i]; r[i + 8] = hh[i]; }
  return r;
}

// B-operand fragment (32x16 f16). lane16 = N column; lanes 0-15 hold K {k..k+7, k+16..k+23},
// lanes 16-31 hold K {k+8..k+15, k+24..k+31} (matches A). `col` indexes [N,K] row-major.
__device__ __forceinline__ v16h load_frag_b(const f16* __restrict__ base, int ld,
                                            int col, int k, int hi) {
  const f16* p = base + (size_t)col * ld + k + hi * 8;
  v8h lo = *(const v8h*)p;
  v8h hh = *(const v8h*)(p + 16);
  v16h r;
#pragma unroll
  for (int i = 0; i < 8; i++) { r[i] = lo[i]; r[i + 8] = hh[i]; }
  return r;
}

// Max reduction across the 16-lane half-wave (C/D layout: one row per half).
__device__ __forceinline__ float half_max(float v) {
  v = fmaxf(v, __shfl_xor(v, 1, 32));
  v = fmaxf(v, __shfl_xor(v, 2, 32));
  v = fmaxf(v, __shfl_xor(v, 4, 32));
  v = fmaxf(v, __shfl_xor(v, 8, 32));
  return v;
}

// ---------------------------------------------------------------------------
__global__ void f32_to_f16(const float* __restrict__ s, f16* __restrict__ d, int n) {
  int i = blockIdx.x * 256 + threadIdx.x;
  if (i < n) d[i] = (f16)s[i];
}

// vt[b][h][d][t] = v part of qkv[b][t][2C + h*D + d]  (coalesced writes)
__global__ void vtrans(const f16* __restrict__ qkv, f16* __restrict__ vt) {
  size_t idx = (size_t)blockIdx.x * 256 + threadIdx.x;   // over B*H*D*T
  int t = (int)(idx % T_);
  size_t r = idx / T_;
  int d = (int)(r % D_);
  size_t r2 = r / D_;
  int h = (int)(r2 % H_);
  int b = (int)(r2 / H_);
  vt[idx] = qkv[((size_t)b * T_ + t) * C3_ + 2 * C_ + h * D_ + d];
}

// out[M,N] = A[M,K] @ Bw[N,K]^T + bias[N]; one wave -> 32x64 tile
// (2 A fragments amortize 4 shared B fragments: 12 b128 loads per 8 WMMA).
template <typename OutT>
__global__ __launch_bounds__(256) void gemm_bias(const f16* __restrict__ A,
                                                 const f16* __restrict__ Bw,
                                                 const float* __restrict__ bias,
                                                 OutT* __restrict__ out,
                                                 int M, int N, int K) {
  int lane   = threadIdx.x & 31;
  int lane16 = lane & 15;
  int hi     = lane >> 4;
  int wid    = blockIdx.x * 8 + (threadIdx.x >> 5);
  int nT     = N >> 6;
  int mt     = wid / nT;
  int n64    = wid % nT;
  int m0     = mt * 32;
  int n0     = n64 * 64;
  if (m0 >= M) return;  // wave-uniform

  v8f acc[2][4] = {};
  for (int k = 0; k < K; k += 32) {
    v16h a0 = load_frag_a(A, K, m0 + lane16,      k, hi);
    v16h a1 = load_frag_a(A, K, m0 + 16 + lane16, k, hi);
#pragma unroll
    for (int t = 0; t < 4; t++) {
      v16h bf = load_frag_b(Bw, K, n0 + t * 16 + lane16, k, hi);
      acc[0][t] = WMMA_F32_F16(a0, bf, acc[0][t]);
      acc[1][t] = WMMA_F32_F16(a1, bf, acc[1][t]);
    }
  }
#pragma unroll
  for (int t = 0; t < 4; t++) {
    float bv = bias[n0 + t * 16 + lane16];
#pragma unroll
    for (int m = 0; m < 2; m++) {
#pragma unroll
      for (int j = 0; j < 8; j++) {
        out[(size_t)(m0 + m * 16 + j + hi * 8) * N + n0 + t * 16 + lane16] =
            (OutT)(acc[m][t][j] + bv);
      }
    }
  }
}

// ---------------------------------------------------------------------------
// One 32-key flash block for a 32-row query tile (two 16-row sub-tiles sharing
// K and V fragments). Row sums of P come from a P @ ones WMMA (no shuffle sum
// tree); only the running-max butterfly uses cross-lane ops.
// MASKED=true only for the single diagonal block. Q pre-scaled by 1/sqrt(D).
template <bool MASKED>
__device__ __forceinline__ void fa_kblock(int s0, int m0, int lane16, int hi,
                                          const f16* __restrict__ kb,
                                          const f16* __restrict__ vb,
                                          f16 (*__restrict__ pb)[16][40],
                                          const v16h (&qa)[2][2],
                                          const v16h& ones,
                                          v8f (&o)[2][4],
                                          float (&mrow)[2][8],
                                          float (&lrow)[2][8]) {
  // Shared K fragments for both query sub-tiles.
  v16h k00 = load_frag_b(kb, C3_, s0 + lane16,      0,  hi);
  v16h k01 = load_frag_b(kb, C3_, s0 + lane16,      32, hi);
  v16h k10 = load_frag_b(kb, C3_, s0 + 16 + lane16, 0,  hi);
  v16h k11 = load_frag_b(kb, C3_, s0 + 16 + lane16, 32, hi);

#pragma unroll
  for (int qt = 0; qt < 2; qt++) {
    // S = Q K^T (two 16x16 tiles, each 2 WMMA over D=64)
    v8f sA = {};
    sA = WMMA_F32_F16(qa[qt][0], k00, sA);
    sA = WMMA_F32_F16(qa[qt][1], k01, sA);
    v8f sB = {};
    sB = WMMA_F32_F16(qa[qt][0], k10, sB);
    sB = WMMA_F32_F16(qa[qt][1], k11, sB);

    // Online softmax max/exp (C-layout: VGPR j is row j + hi*8, col lane16)
#pragma unroll
    for (int j = 0; j < 8; j++) {
      float x0 = sA[j];
      float x1 = sB[j];
      if (MASKED) {
        int row = m0 + qt * 16 + j + hi * 8;
        if (s0 + lane16      > row) x0 = -1.0e30f;
        if (s0 + 16 + lane16 > row) x1 = -1.0e30f;
      }
      float mx = half_max(fmaxf(x0, x1));
      float mn = fmaxf(mrow[qt][j], mx);
      float p0 = __expf(x0 - mn);
      float p1 = __expf(x1 - mn);
      float corr = __expf(mrow[qt][j] - mn);
      mrow[qt][j] = mn;
      lrow[qt][j] *= corr;                       // row sum added after WMMA
      o[qt][0][j] *= corr; o[qt][1][j] *= corr;
      o[qt][2][j] *= corr; o[qt][3][j] *= corr;
      // stage P (f16) into LDS for the A-layout reload
      pb[qt][j + hi * 8][lane16]      = (f16)p0;
      pb[qt][j + hi * 8][16 + lane16] = (f16)p1;
    }
  }
  asm volatile("s_wait_dscnt 0" ::: "memory");

  // Reload both P tiles as A-operand fragments (interleaved K-chunk layout).
  v16h pf[2];
#pragma unroll
  for (int qt = 0; qt < 2; qt++) {
    const f16* pr = &pb[qt][lane16][0];
    v8h plo = *(const v8h*)(pr + hi * 8);
    v8h phi = *(const v8h*)(pr + 16 + hi * 8);
#pragma unroll
    for (int i = 0; i < 8; i++) { pf[qt][i] = plo[i]; pf[qt][i + 8] = phi[i]; }
  }

  // Row sums via matrix pipe: P(16x32) @ ones(32x16) replicates each row's
  // sum across its 16-lane half -- exactly the lrow layout.
#pragma unroll
  for (int qt = 0; qt < 2; qt++) {
    v8f rs = {};
    rs = WMMA_F32_F16(pf[qt], ones, rs);
#pragma unroll
    for (int j = 0; j < 8; j++) lrow[qt][j] += rs[j];
  }

  // O += P @ V: shared V fragments feed both query sub-tiles.
#pragma unroll
  for (int nt = 0; nt < 4; nt++) {
    v16h vf = load_frag_b(vb, T_, nt * 16 + lane16, s0, hi);
    o[0][nt] = WMMA_F32_F16(pf[0], vf, o[0][nt]);
    o[1][nt] = WMMA_F32_F16(pf[1], vf, o[1][nt]);
  }
}

// Flash attention: one wave per (b, h, 32-row query tile), D = 64.
__global__ __launch_bounds__(128, 1) void flash_attn(const f16* __restrict__ qkv,
                                                     const f16* __restrict__ vt,
                                                     f16* __restrict__ oh) {
  __shared__ f16 pbuf[4][2][16][40];  // per-wave P staging (row stride 80B)

  int lane   = threadIdx.x & 31;
  int wib    = threadIdx.x >> 5;      // 0..3
  int lane16 = lane & 15;
  int hi     = lane >> 4;

  int wid = blockIdx.x * 4 + wib;
  const int mtPerHead = T_ / 32;      // 64
  int b  = wid / (H_ * mtPerHead);
  int rr = wid % (H_ * mtPerHead);
  int h  = rr / mtPerHead;
  int m0 = (rr % mtPerHead) * 32;

  const f16* qb = qkv + (size_t)b * T_ * C3_ + h * D_;         // Q rows, stride C3_
  const f16* kb = qkv + (size_t)b * T_ * C3_ + C_ + h * D_;    // K rows, stride C3_
  const f16* vb = vt  + (size_t)(b * H_ + h) * D_ * T_;        // V^T rows, stride T_
  f16 (*pb)[16][40] = pbuf[wib];

  // Q fragments held for whole loop; fold 1/sqrt(D)=0.125 in once (exact pow2)
  v16h qa[2][2];
  qa[0][0] = load_frag_a(qb, C3_, m0 + lane16,      0,  hi);
  qa[0][1] = load_frag_a(qb, C3_, m0 + lane16,      32, hi);
  qa[1][0] = load_frag_a(qb, C3_, m0 + 16 + lane16, 0,  hi);
  qa[1][1] = load_frag_a(qb, C3_, m0 + 16 + lane16, 32, hi);
  const f16 qs = (f16)0.125f;
#pragma unroll
  for (int qt = 0; qt < 2; qt++)
#pragma unroll
    for (int i = 0; i < 16; i++) { qa[qt][0][i] *= qs; qa[qt][1][i] *= qs; }

  v16h ones;
#pragma unroll
  for (int i = 0; i < 16; i++) ones[i] = (f16)1.0f;

  v8f o[2][4] = {};
  float mrow[2][8], lrow[2][8];
#pragma unroll
  for (int qt = 0; qt < 2; qt++)
#pragma unroll
    for (int j = 0; j < 8; j++) { mrow[qt][j] = -3.0e38f; lrow[qt][j] = 0.0f; }

  int sEnd = (m0 + 32 + 31) & ~31;    // = m0 + 32 (m0 multiple of 32)

  // All blocks except the last are strictly below the diagonal: no masking.
  int s0 = 0;
  for (; s0 < sEnd - 32; s0 += 32)
    fa_kblock<false>(s0, m0, lane16, hi, kb, vb, pb, qa, ones, o, mrow, lrow);
  // Diagonal block with causal mask.
  fa_kblock<true>(s0, m0, lane16, hi, kb, vb, pb, qa, ones, o, mrow, lrow);

  f16* ob = oh + ((size_t)b * T_ + m0) * C_ + h * D_;
#pragma unroll
  for (int qt = 0; qt < 2; qt++) {
#pragma unroll
    for (int j = 0; j < 8; j++) {
      float inv = 1.0f / lrow[qt][j];
      size_t rbase = (size_t)(qt * 16 + j + hi * 8) * C_ + lane16;
      ob[rbase + 0]  = (f16)(o[qt][0][j] * inv);
      ob[rbase + 16] = (f16)(o[qt][1][j] * inv);
      ob[rbase + 32] = (f16)(o[qt][2][j] * inv);
      ob[rbase + 48] = (f16)(o[qt][3][j] * inv);
    }
  }
}

// ---------------------------------------------------------------------------
extern "C" void kernel_launch(void* const* d_in, const int* in_sizes, int n_in,
                              void* d_out, int out_size, void* d_ws, size_t ws_size,
                              hipStream_t stream) {
  const float* x      = (const float*)d_in[0];
  const float* attn_w = (const float*)d_in[1];
  const float* attn_b = (const float*)d_in[2];
  const float* proj_w = (const float*)d_in[3];
  const float* proj_b = (const float*)d_in[4];
  float* out = (float*)d_out;

  char* ws = (char*)d_ws;
  size_t off = 0;
  auto alloc = [&](size_t bytes) -> void* {
    void* p = ws + off;
    off += (bytes + 255) & ~(size_t)255;
    return p;
  };
  f16* xh    = (f16*)alloc((size_t)B_ * T_ * C_ * 2);      // x in f16
  f16* wqkvh = (f16*)alloc((size_t)C3_ * C_ * 2);          // Wqkv in f16
  f16* pwh   = (f16*)alloc((size_t)C_ * C_ * 2);           // Wproj in f16
  f16* qkvh  = (f16*)alloc((size_t)B_ * T_ * C3_ * 2);     // qkv in f16
  f16* vth   = (f16*)alloc((size_t)B_ * H_ * D_ * T_ * 2); // V transposed per head
  f16* ohh   = (f16*)alloc((size_t)B_ * T_ * C_ * 2);      // attention out in f16

  int nx = B_ * T_ * C_;
  f32_to_f16<<<nx / 256, 256, 0, stream>>>(x, xh, nx);
  int nw = C3_ * C_;
  f32_to_f16<<<(nw + 255) / 256, 256, 0, stream>>>(attn_w, wqkvh, nw);
  int np = C_ * C_;
  f32_to_f16<<<(np + 255) / 256, 256, 0, stream>>>(proj_w, pwh, np);

  {  // QKV projection: [4096,1024] x [3072,1024]^T
    int M = B_ * T_, N = C3_, K = C_;
    int waves = (M / 32) * (N / 64);
    gemm_bias<_Float16><<<waves / 8, 256, 0, stream>>>(xh, wqkvh, attn_b, qkvh, M, N, K);
  }

  int nv = B_ * H_ * D_ * T_;
  vtrans<<<nv / 256, 256, 0, stream>>>(qkvh, vth);

  int fwaves = B_ * H_ * (T_ / 32);                        // 2048 waves
  flash_attn<<<fwaves / 4, 128, 0, stream>>>(qkvh, vth, ohh);

  {  // output projection: [4096,1024] x [1024,1024]^T -> f32 + bias
    int M = B_ * T_, N = C_, K = C_;
    int waves = (M / 32) * (N / 64);
    gemm_bias<float><<<waves / 8, 256, 0, stream>>>(ohh, pwh, proj_b, out, M, N, K);
  }
}
